// NBSampler_77086073029141
// MI455X (gfx1250) — compile-verified, matrix operand order checked by reading, not run on hardware
//
#include <hip/hip_runtime.h>
#include <math.h>

// Problem constants (from reference)
#define DIM   4096
#define BSZ   256
#define MAXR  19

// GEMM tiling
#define KB 32     // K tile (== WMMA K)
#define NB 32     // N tile per block (2 WMMA N-subtiles per wave)
#define MB 128    // M rows per block (8 waves x 16 rows)

typedef _Float16 v16h __attribute__((ext_vector_type(16)));
typedef _Float16 h8v  __attribute__((ext_vector_type(8)));
typedef float    v8f  __attribute__((ext_vector_type(8)));

// ---------------------------------------------------------------------------
// G = X @ sym(W),  sym(W)[k][n] = 0.5*(W[k][n] + W[n][k])
// X: (BSZ, DIM) f32, W: (DIM, DIM) f32, G: (BSZ, DIM) f32
// grid: (DIM/NB, BSZ/MB), block: 256 threads (8 waves)
//
// The W[n][k] tile is staged into LDS with GLOBAL_LOAD_ASYNC_TO_LDS_B128
// (ASYNCcnt path), double-buffered and issued one K-step ahead so the DMA
// overlaps the sym-tile build + WMMAs of the current step.
// ---------------------------------------------------------------------------
__global__ void __launch_bounds__(256)
gemm_sym_kernel(const float* __restrict__ X, const float* __restrict__ W,
                float* __restrict__ G) {
  __shared__ float    WtS[2][NB][KB];    // W[n][k] staging (f32), double-buffered
  __shared__ _Float16 Bs2[NB][KB];       // sym tile, fragment-major: [n][k]

  const int n0   = blockIdx.x * NB;
  const int m0   = blockIdx.y * MB;
  const int tid  = threadIdx.x;
  const int lane = tid & 31;
  const int wave = tid >> 5;
  const int l16  = lane & 15;
  const bool lo  = (lane & 16) == 0;     // lane group 0-15 vs 16-31

  const int mrow = m0 + wave * 16 + l16; // A-matrix row owned by this lane

  v8f acc0, acc1;
  #pragma unroll
  for (int i = 0; i < 8; ++i) { acc0[i] = 0.0f; acc1[i] = 0.0f; }

  const int ldr  = tid >> 3;             // 0..31 staging row (n)
  const int ldc4 = (tid & 7) * 4;        // 0..28 staging col (float4 chunk, k)

  // async copy of one 32x32 f32 tile: 256 threads x 16B
  auto async_wt_tile = [&](int k0, int buf) {
    const unsigned lws = (unsigned)(uintptr_t)&WtS[buf][ldr][ldc4];
    const unsigned long long gsrc =
        (unsigned long long)(uintptr_t)(W + (size_t)(n0 + ldr) * DIM + k0 + ldc4);
    asm volatile("global_load_async_to_lds_b128 %0, %1, off"
                 :: "v"(lws), "v"(gsrc) : "memory");
  };

  // prologue: fetch tile 0
  async_wt_tile(0, 0);
  asm volatile("s_wait_asynccnt 0x0" ::: "memory");
  __syncthreads();

  for (int k0 = 0; k0 < DIM; k0 += KB) {
    const int buf = (k0 >> 5) & 1;
    // issue DMA for the next K-step's transpose tile (overlaps with below)
    if (k0 + KB < DIM) {
      async_wt_tile(k0 + KB, buf ^ 1);
      __builtin_prefetch(W + (size_t)(k0 + KB + ldr) * DIM + n0, 0, 1);
    }

    // ---- build sym f16 tile: Bs2[n][k] = 0.5*(W[k0+k][n0+n] + W[n0+n][k0+k]) ----
    {
      const float4 w = *(const float4*)(W + (size_t)(k0 + ldr) * DIM + n0 + ldc4);
      Bs2[ldc4 + 0][ldr] = (_Float16)(0.5f * (w.x + WtS[buf][ldc4 + 0][ldr]));
      Bs2[ldc4 + 1][ldr] = (_Float16)(0.5f * (w.y + WtS[buf][ldc4 + 1][ldr]));
      Bs2[ldc4 + 2][ldr] = (_Float16)(0.5f * (w.z + WtS[buf][ldc4 + 2][ldr]));
      Bs2[ldc4 + 3][ldr] = (_Float16)(0.5f * (w.w + WtS[buf][ldc4 + 3][ldr]));
    }
    __syncthreads();

    // ---- A fragment (16x32 f16): lane L holds M=l16, K set per ISA layout:
    //      lo lanes: K = 0..7, 16..23 ; hi lanes: K = 8..15, 24..31 ----
    const float4* xv = (const float4*)(X + (size_t)mrow * DIM + k0);
    const int cb = lo ? 0 : 2;
    const float4 q0 = xv[cb + 0], q1 = xv[cb + 1];
    const float4 q2 = xv[cb + 4], q3 = xv[cb + 5];
    v16h a;
    a[0]=(_Float16)q0.x; a[1]=(_Float16)q0.y; a[2]=(_Float16)q0.z; a[3]=(_Float16)q0.w;
    a[4]=(_Float16)q1.x; a[5]=(_Float16)q1.y; a[6]=(_Float16)q1.z; a[7]=(_Float16)q1.w;
    a[8]=(_Float16)q2.x; a[9]=(_Float16)q2.y; a[10]=(_Float16)q2.z; a[11]=(_Float16)q2.w;
    a[12]=(_Float16)q3.x; a[13]=(_Float16)q3.y; a[14]=(_Float16)q3.z; a[15]=(_Float16)q3.w;

    // ---- B fragments (32x16 f16), mirrored K pattern, N = l16 (+16) ----
    const int kb = lo ? 0 : 8;
    const h8v b0l = *(const h8v*)&Bs2[l16][kb];
    const h8v b0h = *(const h8v*)&Bs2[l16][kb + 16];
    const h8v b1l = *(const h8v*)&Bs2[l16 + 16][kb];
    const h8v b1h = *(const h8v*)&Bs2[l16 + 16][kb + 16];
    v16h b0, b1;
    #pragma unroll
    for (int h = 0; h < 8; ++h) {
      b0[h] = b0l[h]; b0[8 + h] = b0h[h];
      b1[h] = b1l[h]; b1[8 + h] = b1h[h];
    }

    acc0 = __builtin_amdgcn_wmma_f32_16x16x32_f16(false, a, false, b0,
                                                  (short)0, acc0, false, false);
    acc1 = __builtin_amdgcn_wmma_f32_16x16x32_f16(false, a, false, b1,
                                                  (short)0, acc1, false, false);

    // next tile's DMA must have landed before we read it; barrier also
    // protects Bs2/WtS reuse across the whole block
    asm volatile("s_wait_asynccnt 0x0" ::: "memory");
    __syncthreads();
  }

  // ---- C/D layout: VGPR v -> M = v (+8 for hi lanes), N = l16 ----
  const int mbase = m0 + wave * 16 + (lo ? 0 : 8);
  #pragma unroll
  for (int v = 0; v < 8; ++v) {
    G[(size_t)(mbase + v) * DIM + n0 + l16]      = acc0[v];
    G[(size_t)(mbase + v) * DIM + n0 + 16 + l16] = acc1[v];
  }
}

// ---------------------------------------------------------------------------
// score[row] = sum_j x_j*(0.5*G_j + b_j);  sc_j = (0.5 - x_j)*(G_j + b_j)
// one block per row
// ---------------------------------------------------------------------------
__global__ void __launch_bounds__(256)
score_change_kernel(const float* __restrict__ X, const float* __restrict__ G,
                    const float* __restrict__ bvec, float* __restrict__ score,
                    float* __restrict__ sc) {
  __shared__ float red[256];
  const int row = blockIdx.x, tid = threadIdx.x;
  const float* xr = X + (size_t)row * DIM;
  const float* gr = G + (size_t)row * DIM;
  float*       sr = sc + (size_t)row * DIM;
  float acc = 0.0f;
  for (int j = tid; j < DIM; j += 256) {
    const float xv = xr[j], gv = gr[j], bv = bvec[j];
    acc += xv * (0.5f * gv + bv);
    sr[j] = (0.5f - xv) * (gv + bv);
  }
  red[tid] = acc; __syncthreads();
  for (int s = 128; s > 0; s >>= 1) {
    if (tid < s) red[tid] += red[tid + s];
    __syncthreads();
  }
  if (tid == 0) score[row] = red[0];
}

// ---------------------------------------------------------------------------
// Sequential Gibbs scan: 19 steps of gumbel-argmax + masked flip.
// one block per row; logits & bits held in registers (16 per thread)
// ---------------------------------------------------------------------------
__global__ void __launch_bounds__(256)
gibbs_scan_kernel(const float* __restrict__ X, const float* __restrict__ scx,
                  const float* __restrict__ gumbel, const int* __restrict__ radius,
                  float* __restrict__ Y, int* __restrict__ index) {
  __shared__ float rv[256];
  __shared__ int   ri[256];
  __shared__ int   widx;
  const int row = blockIdx.x, tid = threadIdx.x;
  const int rad = radius[row];
  float logits[16], bits[16];
  #pragma unroll
  for (int i = 0; i < 16; ++i) {
    const int j = i * 256 + tid;
    logits[i] = scx[(size_t)row * DIM + j];
    bits[i]   = X[(size_t)row * DIM + j];
  }
  for (int t = 0; t < MAXR; ++t) {
    const float* g = gumbel + ((size_t)t * BSZ + row) * DIM;
    float best = -__builtin_inff(); int bidx = DIM;
    #pragma unroll
    for (int i = 0; i < 16; ++i) {
      const int j = i * 256 + tid;
      const float v = logits[i] + g[j];
      if (v > best || (v == best && j < bidx)) { best = v; bidx = j; }
    }
    rv[tid] = best; ri[tid] = bidx; __syncthreads();
    for (int s = 128; s > 0; s >>= 1) {
      if (tid < s) {
        const float v = rv[tid + s]; const int j = ri[tid + s];
        if (v > rv[tid] || (v == rv[tid] && j < ri[tid])) { rv[tid] = v; ri[tid] = j; }
      }
      __syncthreads();
    }
    if (tid == 0) { widx = ri[0]; index[row * MAXR + t] = ri[0]; }
    __syncthreads();
    const int w = widx;
    if ((w & 255) == tid) {
      const int iw = w >> 8;
      #pragma unroll
      for (int i = 0; i < 16; ++i) if (i == iw) {
        if (t < rad) bits[i] = 1.0f - bits[i];
        logits[i] = -__builtin_inff();
      }
    }
    __syncthreads();
  }
  #pragma unroll
  for (int i = 0; i < 16; ++i)
    Y[(size_t)row * DIM + i * 256 + tid] = bits[i];
}

// ---------------------------------------------------------------------------
// Forward path log-probs: iterate t ascending, mask after use.
// log_fwd[row] = sum_{t<rad} (sc[index_t] - logsumexp(masked sc)) + score_x
// ---------------------------------------------------------------------------
__global__ void __launch_bounds__(256)
logfwd_kernel(const float* __restrict__ scx, const int* __restrict__ index,
              const int* __restrict__ radius, const float* __restrict__ score_x,
              float* __restrict__ log_fwd) {
  __shared__ float red[256];
  __shared__ float sh_target;
  const int row = blockIdx.x, tid = threadIdx.x;
  const int rad = radius[row];
  float l[16];
  #pragma unroll
  for (int i = 0; i < 16; ++i) l[i] = scx[(size_t)row * DIM + i * 256 + tid];
  float acc = 0.0f;
  for (int t = 0; t < MAXR; ++t) {
    const int idx = index[row * MAXR + t];
    const int it = idx & 255, ii = idx >> 8;
    if (tid == it) {
      #pragma unroll
      for (int i = 0; i < 16; ++i) if (i == ii) sh_target = l[i];
    }
    float m = -__builtin_inff();
    #pragma unroll
    for (int i = 0; i < 16; ++i) m = fmaxf(m, l[i]);
    red[tid] = m; __syncthreads();
    for (int s = 128; s > 0; s >>= 1) {
      if (tid < s) red[tid] = fmaxf(red[tid], red[tid + s]);
      __syncthreads();
    }
    const float mx = red[0]; __syncthreads();
    float se = 0.0f;
    #pragma unroll
    for (int i = 0; i < 16; ++i) se += expf(l[i] - mx);
    red[tid] = se; __syncthreads();
    for (int s = 128; s > 0; s >>= 1) {
      if (tid < s) red[tid] += red[tid + s];
      __syncthreads();
    }
    const float lse = mx + logf(red[0]);
    if (t < rad) acc += sh_target - lse;
    if (tid == it) {
      #pragma unroll
      for (int i = 0; i < 16; ++i) if (i == ii) l[i] = -__builtin_inff();
    }
    __syncthreads();
  }
  if (tid == 0) log_fwd[row] = acc + score_x[row];
}

// ---------------------------------------------------------------------------
// Backward path log-probs (t descending, mask set = indices > t) + accept test.
// ---------------------------------------------------------------------------
__global__ void __launch_bounds__(256)
logbwd_accept_kernel(const float* __restrict__ scy, const int* __restrict__ index,
                     const int* __restrict__ radius, const float* __restrict__ score_y,
                     const float* __restrict__ log_fwd, const float* __restrict__ u,
                     float* __restrict__ acc_flag) {
  __shared__ float red[256];
  __shared__ float sh_target;
  const int row = blockIdx.x, tid = threadIdx.x;
  const int rad = radius[row];
  float l[16];
  #pragma unroll
  for (int i = 0; i < 16; ++i) l[i] = scy[(size_t)row * DIM + i * 256 + tid];
  float acc = 0.0f;
  for (int t = MAXR - 1; t >= 0; --t) {
    const int idx = index[row * MAXR + t];
    const int it = idx & 255, ii = idx >> 8;
    if (tid == it) {
      #pragma unroll
      for (int i = 0; i < 16; ++i) if (i == ii) sh_target = l[i];
    }
    float m = -__builtin_inff();
    #pragma unroll
    for (int i = 0; i < 16; ++i) m = fmaxf(m, l[i]);
    red[tid] = m; __syncthreads();
    for (int s = 128; s > 0; s >>= 1) {
      if (tid < s) red[tid] = fmaxf(red[tid], red[tid + s]);
      __syncthreads();
    }
    const float mx = red[0]; __syncthreads();
    float se = 0.0f;
    #pragma unroll
    for (int i = 0; i < 16; ++i) se += expf(l[i] - mx);
    red[tid] = se; __syncthreads();
    for (int s = 128; s > 0; s >>= 1) {
      if (tid < s) red[tid] += red[tid + s];
      __syncthreads();
    }
    const float lse = mx + logf(red[0]);
    if (t < rad) acc += sh_target - lse;
    if (tid == it) {
      #pragma unroll
      for (int i = 0; i < 16; ++i) if (i == ii) l[i] = -__builtin_inff();
    }
    __syncthreads();
  }
  if (tid == 0) {
    const float log_bwd = acc + score_y[row];
    const float log_acc = log_bwd - log_fwd[row];
    acc_flag[row] = (expf(log_acc) >= u[row]) ? 1.0f : 0.0f;
  }
}

// ---------------------------------------------------------------------------
// out = accepted ? y : x
// ---------------------------------------------------------------------------
__global__ void __launch_bounds__(256)
select_kernel(const float* __restrict__ X, const float* __restrict__ Y,
              const float* __restrict__ flag, float* __restrict__ out) {
  const int row = blockIdx.y;
  const int col = blockIdx.x * 256 + threadIdx.x;
  const size_t o = (size_t)row * DIM + col;
  out[o] = (flag[row] > 0.5f) ? Y[o] : X[o];
}

// ---------------------------------------------------------------------------
extern "C" void kernel_launch(void* const* d_in, const int* in_sizes, int n_in,
                              void* d_out, int out_size, void* d_ws, size_t ws_size,
                              hipStream_t stream) {
  (void)in_sizes; (void)n_in; (void)out_size; (void)ws_size;
  const float* x      = (const float*)d_in[0];
  const float* W      = (const float*)d_in[1];
  const float* bvec   = (const float*)d_in[2];
  const int*   radius = (const int*)d_in[3];
  const float* gumbel = (const float*)d_in[4];
  const float* u      = (const float*)d_in[5];
  float* out = (float*)d_out;

  // workspace layout (~20 MB)
  float* Gx       = (float*)d_ws;
  float* Gy       = Gx + (size_t)BSZ * DIM;
  float* yb       = Gy + (size_t)BSZ * DIM;
  float* scx      = yb + (size_t)BSZ * DIM;
  float* scy      = scx + (size_t)BSZ * DIM;
  float* score_x  = scy + (size_t)BSZ * DIM;
  float* score_y  = score_x + BSZ;
  float* log_fwd  = score_y + BSZ;
  float* acc_flag = log_fwd + BSZ;
  int*   index    = (int*)(acc_flag + BSZ);

  const dim3 ggrid(DIM / NB, BSZ / MB);

  gemm_sym_kernel<<<ggrid, 256, 0, stream>>>(x, W, Gx);
  score_change_kernel<<<BSZ, 256, 0, stream>>>(x, Gx, bvec, score_x, scx);
  gibbs_scan_kernel<<<BSZ, 256, 0, stream>>>(x, scx, gumbel, radius, yb, index);
  gemm_sym_kernel<<<ggrid, 256, 0, stream>>>(yb, W, Gy);
  score_change_kernel<<<BSZ, 256, 0, stream>>>(yb, Gy, bvec, score_y, scy);
  logfwd_kernel<<<BSZ, 256, 0, stream>>>(scx, index, radius, score_x, log_fwd);
  logbwd_accept_kernel<<<BSZ, 256, 0, stream>>>(scy, index, radius, score_y,
                                                log_fwd, u, acc_flag);
  select_kernel<<<dim3(DIM / 256, BSZ), 256, 0, stream>>>(x, yb, acc_flag, out);
}
